// RNN_3805341024632
// MI455X (gfx1250) — matrix operand order repaired and checked by reading, not compiled, not run on hardware
//
#include <hip/hip_runtime.h>
#include <hip/hip_bf16.h>
#include <math.h>

// ---------------------------------------------------------------------------
// Problem dims (fixed by the reference)
// ---------------------------------------------------------------------------
#define B_  64
#define L_  512
#define I_  256
#define H_  1024
#define M_  (B_ * L_)          // 32768 rows in the bulk GEMMs

typedef __attribute__((ext_vector_type(16))) __bf16 v16bf;
typedef __attribute__((ext_vector_type(8)))  __bf16 v8bf;
typedef __attribute__((ext_vector_type(8)))  float  v8f;
typedef __attribute__((ext_vector_type(4)))  int    v4i;

#define AS1 __attribute__((address_space(1)))
#define AS3 __attribute__((address_space(3)))

// ---------------------------------------------------------------------------
// gfx1250 async global->LDS path (ASYNCcnt-tracked), with safe fallback.
// Probe result: __builtin_amdgcn_global_load_async_to_lds_b128 takes
// (AS1 int4*, AS3 int4*, imm offset, imm cpol).
// ---------------------------------------------------------------------------
#if defined(__AMDGCN__) && __has_builtin(__builtin_amdgcn_global_load_async_to_lds_b128)
#define HAVE_ASYNC_LDS 1
#else
#define HAVE_ASYNC_LDS 0
#endif

#if HAVE_ASYNC_LDS
#if __has_builtin(__builtin_amdgcn_s_wait_asynccnt)
#define WAIT_ASYNC(n) __builtin_amdgcn_s_wait_asynccnt(n)
#else
#define WAIT_ASYNC(n) asm volatile("s_wait_asynccnt %0" ::"i"(n) : "memory")
#endif
#else
#define WAIT_ASYNC(n) ((void)0)
#endif

// Copy 8 bf16 (16 B) from global to LDS.
__device__ __forceinline__ void g2l_16B(const __bf16* __restrict__ g,
                                        __bf16* __restrict__ l) {
#if HAVE_ASYNC_LDS
  void* gp = (void*)g;   // strip const, then cast generic -> AS1/AS3 typed ptrs
  void* lp = (void*)l;
  __builtin_amdgcn_global_load_async_to_lds_b128((AS1 v4i*)gp, (AS3 v4i*)lp, 0, 0);
#else
  *(v8bf*)l = *(const v8bf*)g;   // sync fallback: global_load + ds_store
#endif
}

// ---------------------------------------------------------------------------
// WMMA tile loaders for C = A * W^T (NT GEMM, K contiguous in both operands).
//
// A (16x32, MxK):  lane l<16 holds row m0+l, K k0..k0+15 (16 consecutive K)
//                  lane l+16 holds row m0+l, K k0+16..k0+31
// B (32x16, KxN)=W^T: lane l<16 holds W[n0+l][k0..k0+15] (16 consecutive K)
//                     lane l+16 holds W[n0+l][k0+16..k0+31]
// ---------------------------------------------------------------------------
__device__ __forceinline__ v16bf load_a_tile(const __bf16* __restrict__ A,
                                             int lda, int m0, int k0, int lane) {
  const int r = lane & 15, half = lane >> 4;
  const __bf16* p = A + (size_t)(m0 + r) * (size_t)lda + (k0 + half * 16);
  union { v16bf v; v8bf h[2]; } u;
  u.h[0] = *(const v8bf*)(p);
  u.h[1] = *(const v8bf*)(p + 8);
  return u.v;
}

__device__ __forceinline__ v16bf load_b_tile_g(const __bf16* __restrict__ W,
                                               int ldb, int n0, int k0, int lane) {
  const int r = lane & 15, half = lane >> 4;
  const __bf16* p = W + (size_t)(n0 + r) * (size_t)ldb + (k0 + half * 16);
  return *(const v16bf*)(p);
}

// B tile from an LDS chunk laid out as [128 rows][32 k] bf16.
__device__ __forceinline__ v16bf load_b_tile_lds(const __bf16* lds_chunk,
                                                 int n_local, int lane) {
  const int r = lane & 15, half = lane >> 4;
  const __bf16* p = lds_chunk + (n_local + r) * 32 + half * 16;
  return *(const v16bf*)(p);
}

// ---------------------------------------------------------------------------
// Bulk GEMM: C[M x N] (f32) = A[M x K] (bf16) @ W[N x K]^T (bf16) + b1[n]+b2[n]
// Block = 256 thr (8 waves). All 8 waves share one 128-col weight slice, so
// W chunks (128 x 32 k = 8 KB) are double-buffered through LDS with async
// copies: next chunk streams in while WMMAs consume the current one.
// Wave -> 16 rows x 128 cols (8 f32 accumulators).
// grid = (M/128, N/128).
// ---------------------------------------------------------------------------
__global__ __launch_bounds__(256) void gemm_bias_bf16_wmma(
    const __bf16* __restrict__ A, const __bf16* __restrict__ W,
    const float* __restrict__ b1, const float* __restrict__ b2,
    float* __restrict__ C, int M, int N, int K)
{
  __shared__ __align__(64) __bf16 lds_b[2][128 * 32];   // 2 x 8 KB

  const int tid  = threadIdx.x;
  const int lane = tid & 31;
  const int wave = tid >> 5;
  const int m0 = (blockIdx.x * 8 + wave) * 16;
  const int nb = blockIdx.y * 128;

  // Per-thread share of one 8 KB chunk: 32 B (2 x 16 B) of one weight row.
  const int c_row  = tid >> 1;                 // 0..127
  const int c_part = (tid & 1) * 16;           // 0 or 16 elements

  // Prologue: stage chunk for k0 = 0 into buffer 0.
  {
    const __bf16* g = W + (size_t)(nb + c_row) * (size_t)K + c_part;
    __bf16* l = &lds_b[0][c_row * 32 + c_part];
    g2l_16B(g, l);
    g2l_16B(g + 8, l + 8);
  }

  v8f zero = {};
  v8f acc[8];
#pragma unroll
  for (int j = 0; j < 8; ++j) acc[j] = zero;

  for (int k0 = 0; k0 < K; k0 += 32) {
    const int cur = (k0 >> 5) & 1;
    if (k0 + 32 < K) {
      // Issue async copies for the NEXT chunk into the other buffer.
      const __bf16* g = W + (size_t)(nb + c_row) * (size_t)K + (k0 + 32) + c_part;
      __bf16* l = &lds_b[cur ^ 1][c_row * 32 + c_part];
      g2l_16B(g, l);
      g2l_16B(g + 8, l + 8);
      WAIT_ASYNC(2);   // wait for CURRENT chunk's 2 copies (in-order), keep next in flight
    } else {
      WAIT_ASYNC(0);
    }
    __syncthreads();   // all waves' copies of `cur` visible

    v16bf a = load_a_tile(A, K, m0, k0, lane);
    if (k0 + 32 < K)   // pull next A chunk toward the WGP (global_prefetch)
      __builtin_prefetch(A + (size_t)(m0 + (lane & 15)) * (size_t)K + k0 + 32, 0, 3);

    v16bf b[8];
#pragma unroll
    for (int j = 0; j < 8; ++j) b[j] = load_b_tile_lds(lds_b[cur], j * 16, lane);
#pragma unroll
    for (int j = 0; j < 8; ++j)
      acc[j] = __builtin_amdgcn_wmma_f32_16x16x32_bf16(
          false, a, false, b[j], (short)0, acc[j], false, false);

    __syncthreads();   // done reading `cur`; it may be overwritten next round
  }

  const int r = lane & 15, half = lane >> 4;
#pragma unroll
  for (int j = 0; j < 8; ++j) {
    const int col = nb + j * 16 + r;
    const float bias = b1[col] + b2[col];
#pragma unroll
    for (int rr = 0; rr < 8; ++rr) {
      const int row = m0 + rr + half * 8;
      C[(size_t)row * (size_t)N + col] = acc[j][rr] + bias;
    }
  }
}

// ---------------------------------------------------------------------------
// One recurrence step: h_next = tanh(xp_t + h_prev @ W_hh^T)
// h_prev/h_next: [64 x 1024] bf16. One wave -> 16 rows x 32 cols (2 accums);
// each 32-col slice of W_hh is read by 4 waves (one per row tile), and the
// matrix is L2-resident, so no LDS/barriers on the latency-critical path.
// 128 waves -> grid = 16 blocks x 256 threads.
// ---------------------------------------------------------------------------
__global__ __launch_bounds__(256) void rnn_step_wmma(
    const __bf16* __restrict__ h_prev,
    const __bf16* __restrict__ Whh,
    const float*  __restrict__ xp_t, int ldx,
    __bf16* __restrict__ h_next,
    __bf16* __restrict__ seq_out)
{
  const int lane = threadIdx.x & 31;
  const int wave = threadIdx.x >> 5;
  const int w  = blockIdx.x * 8 + wave;   // 0..127
  const int m0 = (w & 3) * 16;            // 4 row tiles cover B=64
  const int nb = (w >> 2) * 32;           // 32 col groups cover H=1024

  v8f zero = {};
  v8f acc[2] = { zero, zero };

  for (int k0 = 0; k0 < H_; k0 += 32) {
    v16bf a = load_a_tile(h_prev, H_, m0, k0, lane);
    v16bf b0 = load_b_tile_g(Whh, H_, nb,      k0, lane);
    v16bf b1 = load_b_tile_g(Whh, H_, nb + 16, k0, lane);
    acc[0] = __builtin_amdgcn_wmma_f32_16x16x32_bf16(
        false, a, false, b0, (short)0, acc[0], false, false);
    acc[1] = __builtin_amdgcn_wmma_f32_16x16x32_bf16(
        false, a, false, b1, (short)0, acc[1], false, false);
  }

  const int r = lane & 15, half = lane >> 4;
#pragma unroll
  for (int j = 0; j < 2; ++j) {
    const int col = nb + j * 16 + r;
#pragma unroll
    for (int rr = 0; rr < 8; ++rr) {
      const int row = m0 + rr + half * 8;                 // batch index
      const float v = acc[j][rr] + xp_t[(size_t)row * (size_t)ldx + col];
      const __bf16 hv = (__bf16)tanhf(v);
      h_next[row * H_ + col] = hv;
      if (seq_out) seq_out[(size_t)row * (size_t)ldx + col] = hv;
    }
  }
}

// ---------------------------------------------------------------------------
// Small helpers
// ---------------------------------------------------------------------------
__global__ void cast_f32_to_bf16(const float* __restrict__ src,
                                 __bf16* __restrict__ dst, int n) {
  int i = blockIdx.x * blockDim.x + threadIdx.x;
  if (i < n) dst[i] = (__bf16)src[i];
}

__global__ void zero_bf16(__bf16* __restrict__ p, int n) {
  int i = blockIdx.x * blockDim.x + threadIdx.x;
  if (i < n) p[i] = (__bf16)0.0f;
}

// out[b] = sigmoid(dot(h[b,:], Wfc) + bfc); one wave per batch row.
__global__ __launch_bounds__(256) void fc_sigmoid(
    const __bf16* __restrict__ h, const float* __restrict__ Wfc,
    const float* __restrict__ bfc, float* __restrict__ out)
{
  const int lane = threadIdx.x & 31;
  const int b = blockIdx.x * 8 + (threadIdx.x >> 5);  // grid 8 x 8 waves = 64
  float s = 0.0f;
  for (int k = lane; k < H_; k += 32)
    s += (float)h[b * H_ + k] * Wfc[k];
#pragma unroll
  for (int off = 16; off > 0; off >>= 1)
    s += __shfl_xor(s, off, 32);
  if (lane == 0) out[b] = 1.0f / (1.0f + __expf(-(s + bfc[0])));
}

// ---------------------------------------------------------------------------
// Workspace layout (all offsets 256B aligned). xp buffer reused across layers.
// ---------------------------------------------------------------------------
#define XP_OFF   ((size_t)0)                              // f32 [M_,H_]  128MB
#define H1_OFF   (XP_OFF  + (size_t)M_ * H_ * 4)          // bf16 [M_,H_]  64MB
#define XBF_OFF  (H1_OFF  + (size_t)M_ * H_ * 2)          // bf16 x        16MB
#define WI0_OFF  (XBF_OFF + (size_t)M_ * I_ * 2)          // bf16 W_ih0   512KB
#define WH0_OFF  (WI0_OFF + (size_t)H_ * I_ * 2)          // bf16 W_hh0     2MB
#define WI1_OFF  (WH0_OFF + (size_t)H_ * H_ * 2)          // bf16 W_ih1     2MB
#define WH1_OFF  (WI1_OFF + (size_t)H_ * H_ * 2)          // bf16 W_hh1     2MB
#define HA_OFF   (WH1_OFF + (size_t)H_ * H_ * 2)          // bf16 h ping  128KB
#define HB_OFF   (HA_OFF  + (size_t)B_ * H_ * 2)          // bf16 h pong  128KB

extern "C" void kernel_launch(void* const* d_in, const int* in_sizes, int n_in,
                              void* d_out, int out_size, void* d_ws, size_t ws_size,
                              hipStream_t stream)
{
  (void)in_sizes; (void)n_in; (void)out_size; (void)ws_size;

  const float* x    = (const float*)d_in[0];
  const float* Wih0 = (const float*)d_in[1];
  const float* Whh0 = (const float*)d_in[2];
  const float* bih0 = (const float*)d_in[3];
  const float* bhh0 = (const float*)d_in[4];
  const float* Wih1 = (const float*)d_in[5];
  const float* Whh1 = (const float*)d_in[6];
  const float* bih1 = (const float*)d_in[7];
  const float* bhh1 = (const float*)d_in[8];
  const float* Wfc  = (const float*)d_in[9];
  const float* bfc  = (const float*)d_in[10];
  float* out = (float*)d_out;

  char* ws = (char*)d_ws;
  float*  xp    = (float*) (ws + XP_OFF);
  __bf16* h1    = (__bf16*)(ws + H1_OFF);
  __bf16* xbf   = (__bf16*)(ws + XBF_OFF);
  __bf16* wih0b = (__bf16*)(ws + WI0_OFF);
  __bf16* whh0b = (__bf16*)(ws + WH0_OFF);
  __bf16* wih1b = (__bf16*)(ws + WI1_OFF);
  __bf16* whh1b = (__bf16*)(ws + WH1_OFF);
  __bf16* hA    = (__bf16*)(ws + HA_OFF);
  __bf16* hB    = (__bf16*)(ws + HB_OFF);

  // ---- cast inputs/weights to bf16 (weights then stay resident in L2) ----
  {
    int n;
    n = M_ * I_; cast_f32_to_bf16<<<(n + 255) / 256, 256, 0, stream>>>(x,    xbf,   n);
    n = H_ * I_; cast_f32_to_bf16<<<(n + 255) / 256, 256, 0, stream>>>(Wih0, wih0b, n);
    n = H_ * H_; cast_f32_to_bf16<<<(n + 255) / 256, 256, 0, stream>>>(Whh0, whh0b, n);
    n = H_ * H_; cast_f32_to_bf16<<<(n + 255) / 256, 256, 0, stream>>>(Wih1, wih1b, n);
    n = H_ * H_; cast_f32_to_bf16<<<(n + 255) / 256, 256, 0, stream>>>(Whh1, whh1b, n);
  }

  const dim3 gemm_grid(M_ / 128, H_ / 128);

  // ---- layer 0: xp0 = x @ W_ih0^T + b_ih0 + b_hh0, then 512 WMMA steps ----
  gemm_bias_bf16_wmma<<<gemm_grid, 256, 0, stream>>>(
      xbf, wih0b, bih0, bhh0, xp, M_, H_, I_);

  zero_bf16<<<(B_ * H_ + 255) / 256, 256, 0, stream>>>(hA, B_ * H_);
  __bf16* cur = hA; __bf16* nxt = hB;
  for (int t = 0; t < L_; ++t) {
    rnn_step_wmma<<<16, 256, 0, stream>>>(
        cur, whh0b, xp + (size_t)t * H_, L_ * H_, nxt, h1 + (size_t)t * H_);
    __bf16* tmp = cur; cur = nxt; nxt = tmp;
  }

  // ---- layer 1: xp1 = h1 @ W_ih1^T + b_ih1 + b_hh1 (xp buffer reused) ----
  gemm_bias_bf16_wmma<<<gemm_grid, 256, 0, stream>>>(
      h1, wih1b, bih1, bhh1, xp, M_, H_, H_);

  zero_bf16<<<(B_ * H_ + 255) / 256, 256, 0, stream>>>(hA, B_ * H_);
  cur = hA; nxt = hB;
  for (int t = 0; t < L_; ++t) {
    rnn_step_wmma<<<16, 256, 0, stream>>>(
        cur, whh1b, xp + (size_t)t * H_, L_ * H_, nxt, (__bf16*)nullptr);
    __bf16* tmp = cur; cur = nxt; nxt = tmp;
  }

  // ---- head: sigmoid(h_last @ W_fc^T + b_fc) -> out[64] ----
  fc_sigmoid<<<8, 256, 0, stream>>>(cur, Wfc, bfc, out);
}
